// MetaAdapterController_51539607552027
// MI455X (gfx1250) — matrix operand order repaired and checked by reading, not compiled, not run on hardware
//
#include <hip/hip_runtime.h>
#include <hip/hip_bf16.h>

// ---------------------------------------------------------------------------
// MetaAdapterController fused kernel for MI455X (gfx1250, wave32, WMMA).
// B=32, S=2048, D=512, A=64, E=64. Memory-bound (~256MB min traffic);
// bf16 WMMA (f32 accumulate) makes the 8.6 GFLOP of adapter GEMMs free.
// Weight staging uses GLOBAL_LOAD_ASYNC_TO_LDS_B128 (ASYNCcnt), bypassing
// VGPRs and overlapping with the pre-LN phase.
// ---------------------------------------------------------------------------

#define B_ 32
#define S_ 2048
#define D_ 512
#define A_ 64
#define E_ 64
#define EPS_ 1e-5f

typedef __bf16 bf16;
typedef __bf16 bf16x8 __attribute__((ext_vector_type(8)));
typedef __bf16 v16bf  __attribute__((ext_vector_type(16)));
typedef float  v8f    __attribute__((ext_vector_type(8)));

#if defined(__has_builtin)
#if __has_builtin(__builtin_amdgcn_global_load_async_to_lds_b128)
#define HAVE_ASYNC_LDS 1
#endif
#endif

#ifdef HAVE_ASYNC_LDS
// Builtin signature (from hipcc diagnostic): param is a pointer to a 16-byte
// int vector; global source is AS(1), LDS destination is AS(3).
typedef int v4i_t __attribute__((vector_size(4 * sizeof(int))));
typedef __attribute__((address_space(1))) v4i_t g_v4i;
typedef __attribute__((address_space(3))) v4i_t l_v4i;
#endif

// ----------------------- Phase 1: hypernet GEMVs ---------------------------
// out[b][col] = sum_e emb[b][e] * W[e][col] + bias[col]
// One thread per column, 32 batch accumulators in registers; emb in LDS.
__global__ __launch_bounds__(256) void hyper_kernel(
    const float* __restrict__ emb, const float* __restrict__ W,
    const float* __restrict__ bias, float* __restrict__ out_f,
    bf16* __restrict__ out_b, int ncols) {
  __shared__ float semb[B_ * E_];
  for (int i = threadIdx.x; i < B_ * E_; i += blockDim.x) semb[i] = emb[i];
  __syncthreads();
  int col = blockIdx.x * blockDim.x + threadIdx.x;
  if (col >= ncols) return;
  float acc[B_];
#pragma unroll
  for (int b = 0; b < B_; ++b) acc[b] = 0.f;
  float bi = bias[col];
#pragma unroll 4
  for (int e = 0; e < E_; ++e) {
    float w = W[(size_t)e * ncols + col];
#pragma unroll
    for (int b = 0; b < B_; ++b) acc[b] += semb[b * E_ + e] * w;
  }
#pragma unroll
  for (int b = 0; b < B_; ++b) {
    float v = acc[b] + bi;
    if (out_f) out_f[(size_t)b * ncols + col] = v;
    else       out_b[(size_t)b * ncols + col] = (bf16)v;
  }
}

// ----------------------- Phase 2: fused adapter ----------------------------
// LDS layout (bytes), padded strides to avoid bank conflicts:
#define WDSTR 520   // bf16, w_down row stride (D + 8)
#define WUSTR 72    // bf16, w_up row stride (A + 8)
#define ZSTR  520   // bf16
#define MIDSTR 72   // bf16
#define OUTSTR 516  // f32
#define OFF_WD  0
#define OFF_WU  66560
#define OFF_Z   140288
#define OFF_MID 173568
#define OFF_OUT 178176
#define SMEM_BYTES 244224
#define TILE_S 32

__device__ __forceinline__ float wave_reduce_add(float v) {
#pragma unroll
  for (int m = 16; m >= 1; m >>= 1) v += __shfl_xor(v, m, 32);
  return v;
}

__global__ __launch_bounds__(256, 1) void adapter_kernel(
    const float* __restrict__ x, const bf16* __restrict__ wd_g,
    const bf16* __restrict__ wu_g, const float* __restrict__ b_down,
    const float* __restrict__ b_up, const float* __restrict__ pre_w,
    const float* __restrict__ pre_b, const float* __restrict__ post_w,
    const float* __restrict__ post_b, float* __restrict__ out) {
  extern __shared__ char smem[];
  bf16*  s_wd  = (bf16*)(smem + OFF_WD);
  bf16*  s_wu  = (bf16*)(smem + OFF_WU);
  bf16*  s_z   = (bf16*)(smem + OFF_Z);
  bf16*  s_mid = (bf16*)(smem + OFF_MID);
  float* s_out = (float*)(smem + OFF_OUT);

  const int b  = blockIdx.y;
  const int s0 = blockIdx.x * TILE_S;
  const int tid  = threadIdx.x;
  const int lane = tid & 31;
  const int wv   = tid >> 5;           // 8 waves

  // ---- Phase 0: stage per-sample bf16 adapter weights into LDS ----
  // Async global->LDS (B128, ASYNCcnt): bypasses VGPRs and overlaps with the
  // pre-LN global reads below.
  {
    const bf16* gwd = wd_g + (size_t)b * A_ * D_;   // 4096 x 16B
    const bf16* gwu = wu_g + (size_t)b * D_ * A_;   // 4096 x 16B
#ifdef HAVE_ASYNC_LDS
    for (int i = tid; i < 4096; i += 256) {
      int row = i >> 6, c = i & 63;                 // 64 x 16B per row
      __builtin_amdgcn_global_load_async_to_lds_b128(
          (g_v4i*)(gwd + i * 8), (l_v4i*)(s_wd + row * WDSTR + c * 8), 0, 0);
    }
    for (int i = tid; i < 4096; i += 256) {
      int row = i >> 3, c = i & 7;                  // 8 x 16B per row
      __builtin_amdgcn_global_load_async_to_lds_b128(
          (g_v4i*)(gwu + i * 8), (l_v4i*)(s_wu + row * WUSTR + c * 8), 0, 0);
    }
#else
    for (int i = tid; i < 4096; i += 256) {
      int row = i >> 6, c = i & 63;
      *(uint4*)(s_wd + row * WDSTR + c * 8) = ((const uint4*)gwd)[i];
    }
    for (int i = tid; i < 4096; i += 256) {
      int row = i >> 3, c = i & 7;
      *(uint4*)(s_wu + row * WUSTR + c * 8) = ((const uint4*)gwu)[i];
    }
#endif
  }

  // ---- Phase 1: pre conditional LayerNorm -> z (bf16 in LDS) ----
  // Each wave owns 4 rows; keep x in registers for the residual later.
  float xsav[4][16];
  float cw[16], cb[16];
#pragma unroll
  for (int j = 0; j < 16; ++j) {
    int col = lane + 32 * j;
    cw[j] = pre_w[b * D_ + col];
    cb[j] = pre_b[b * D_ + col];
  }
#pragma unroll
  for (int i = 0; i < 4; ++i) {
    int r = wv * 4 + i;
    const float* xr = x + ((size_t)b * S_ + s0 + r) * D_;
    float s = 0.f, s2 = 0.f;
#pragma unroll
    for (int j = 0; j < 16; ++j) {
      float v = xr[lane + 32 * j];
      xsav[i][j] = v;
      s += v; s2 += v * v;
    }
    s = wave_reduce_add(s); s2 = wave_reduce_add(s2);
    float mean = s * (1.f / D_);
    float rstd = rsqrtf(s2 * (1.f / D_) - mean * mean + EPS_);
#pragma unroll
    for (int j = 0; j < 16; ++j) {
      float zv = (xsav[i][j] - mean) * rstd * cw[j] + cb[j];
      s_z[r * ZSTR + lane + 32 * j] = (bf16)zv;
    }
  }
#ifdef HAVE_ASYNC_LDS
  // All of this wave's async weight transfers must land before the barrier.
#if __has_builtin(__builtin_amdgcn_s_wait_asynccnt)
  __builtin_amdgcn_s_wait_asynccnt(0);
#else
  asm volatile("s_wait_asynccnt 0" ::: "memory");
#endif
#endif
  __syncthreads();

  // ---- Phase 2: down GEMM  (32xD) x (AxD)^T -> mid (32xA), ReLU ----
  // Tile grid: 2 m-tiles x 4 n-tiles = 8 tiles, one per wave. K = 512.
  {
    const int mt = wv & 1, nt = wv >> 1;
    const int arow = mt * 16 + (lane & 15);
    const int hi8  = (lane >> 4) << 3;   // 0 or 8  (A k-offset)
    const int hi16 = (lane >> 4) << 3;   // 0 or 8  (B k-offset)
    const int bcol = nt * 16 + (lane & 15);
    v8f acc = {0.f, 0.f, 0.f, 0.f, 0.f, 0.f, 0.f, 0.f};
#pragma unroll 4
    for (int k0 = 0; k0 < D_; k0 += 32) {
      bf16x8 alo = *(const bf16x8*)(s_z + arow * ZSTR + k0 + hi8);
      bf16x8 ahi = *(const bf16x8*)(s_z + arow * ZSTR + k0 + hi8 + 16);
      v16bf af = __builtin_shufflevector(alo, ahi, 0, 1, 2, 3, 4, 5, 6, 7,
                                         8, 9, 10, 11, 12, 13, 14, 15);
      bf16x8 blo = *(const bf16x8*)(s_wd + bcol * WDSTR + k0 + hi16);
      bf16x8 bhi = *(const bf16x8*)(s_wd + bcol * WDSTR + k0 + hi16 + 16);
      v16bf bf = __builtin_shufflevector(blo, bhi, 0, 1, 2, 3, 4, 5, 6, 7,
                                         8, 9, 10, 11, 12, 13, 14, 15);
      acc = __builtin_amdgcn_wmma_f32_16x16x32_bf16(
          false, af, false, bf, (short)0, acc, false, false);
    }
    float bd = b_down[b * A_ + bcol];
#pragma unroll
    for (int v = 0; v < 8; ++v) {
      int m = mt * 16 + v + hi8;
      float mv = acc[v] + bd;
      s_mid[m * MIDSTR + bcol] = (bf16)(mv > 0.f ? mv : 0.f);
    }
  }
  __syncthreads();

  // ---- Phase 3: up GEMM  (32xA) x (DxA)^T -> out (32xD) + b_up ----
  // 2 m-tiles x 32 n-tiles = 64 tiles; each wave does 8. K = 64.
  {
    const int mt = wv & 1, grp = wv >> 1;
    const int arow = mt * 16 + (lane & 15);
    const int hi8  = (lane >> 4) << 3;
    const int hi16 = (lane >> 4) << 3;
#pragma unroll
    for (int j = 0; j < 8; ++j) {
      const int n0 = (grp * 8 + j) * 16;
      const int bcol = n0 + (lane & 15);
      v8f acc = {0.f, 0.f, 0.f, 0.f, 0.f, 0.f, 0.f, 0.f};
#pragma unroll
      for (int k0 = 0; k0 < A_; k0 += 32) {
        bf16x8 alo = *(const bf16x8*)(s_mid + arow * MIDSTR + k0 + hi8);
        bf16x8 ahi = *(const bf16x8*)(s_mid + arow * MIDSTR + k0 + hi8 + 16);
        v16bf af = __builtin_shufflevector(alo, ahi, 0, 1, 2, 3, 4, 5, 6, 7,
                                           8, 9, 10, 11, 12, 13, 14, 15);
        bf16x8 blo = *(const bf16x8*)(s_wu + bcol * WUSTR + k0 + hi16);
        bf16x8 bhi = *(const bf16x8*)(s_wu + bcol * WUSTR + k0 + hi16 + 16);
        v16bf bf = __builtin_shufflevector(blo, bhi, 0, 1, 2, 3, 4, 5, 6, 7,
                                           8, 9, 10, 11, 12, 13, 14, 15);
        acc = __builtin_amdgcn_wmma_f32_16x16x32_bf16(
            false, af, false, bf, (short)0, acc, false, false);
      }
      float bu = b_up[b * D_ + bcol];
#pragma unroll
      for (int v = 0; v < 8; ++v) {
        int m = mt * 16 + v + hi8;
        s_out[m * OUTSTR + bcol] = acc[v] + bu;
      }
    }
  }
  __syncthreads();

  // ---- Phase 4: post conditional LayerNorm + residual -> global ----
#pragma unroll
  for (int j = 0; j < 16; ++j) {
    int col = lane + 32 * j;
    cw[j] = post_w[b * D_ + col];
    cb[j] = post_b[b * D_ + col];
  }
#pragma unroll
  for (int i = 0; i < 4; ++i) {
    int r = wv * 4 + i;
    float o[16];
    float s = 0.f, s2 = 0.f;
#pragma unroll
    for (int j = 0; j < 16; ++j) {
      float v = s_out[r * OUTSTR + lane + 32 * j];
      o[j] = v; s += v; s2 += v * v;
    }
    s = wave_reduce_add(s); s2 = wave_reduce_add(s2);
    float mean = s * (1.f / D_);
    float rstd = rsqrtf(s2 * (1.f / D_) - mean * mean + EPS_);
    float* orow = out + ((size_t)b * S_ + s0 + r) * D_;
#pragma unroll
    for (int j = 0; j < 16; ++j) {
      orow[lane + 32 * j] =
          (o[j] - mean) * rstd * cw[j] + cb[j] + xsav[i][j];
    }
  }
}

// ---------------------------------------------------------------------------
extern "C" void kernel_launch(void* const* d_in, const int* in_sizes, int n_in,
                              void* d_out, int out_size, void* d_ws,
                              size_t ws_size, hipStream_t stream) {
  const float* emb      = (const float*)d_in[0];
  const float* x        = (const float*)d_in[1];
  const float* down_w_W = (const float*)d_in[2];
  const float* down_w_b = (const float*)d_in[3];
  const float* down_b_W = (const float*)d_in[4];
  const float* down_b_b = (const float*)d_in[5];
  const float* up_w_W   = (const float*)d_in[6];
  const float* up_w_b   = (const float*)d_in[7];
  const float* up_b_W   = (const float*)d_in[8];
  const float* up_b_b   = (const float*)d_in[9];
  const float* pre_w_W  = (const float*)d_in[10];
  const float* pre_w_b  = (const float*)d_in[11];
  const float* pre_b_W  = (const float*)d_in[12];
  const float* pre_b_b  = (const float*)d_in[13];
  const float* post_w_W = (const float*)d_in[14];
  const float* post_w_b = (const float*)d_in[15];
  const float* post_b_W = (const float*)d_in[16];
  const float* post_b_b = (const float*)d_in[17];

  char* ws = (char*)d_ws;
  bf16*  ws_wd    = (bf16*)(ws + 0);                       // 2 MB
  bf16*  ws_wu    = (bf16*)(ws + (2u << 20));              // 2 MB
  float* ws_bdown = (float*)(ws + (4u << 20));             // 8 KB
  float* ws_bup   = (float*)(ws + (4u << 20) + 8192);      // 64 KB
  float* ws_prew  = (float*)(ws + (4u << 20) + 8192 + 1 * 65536);
  float* ws_preb  = (float*)(ws + (4u << 20) + 8192 + 2 * 65536);
  float* ws_postw = (float*)(ws + (4u << 20) + 8192 + 3 * 65536);
  float* ws_postb = (float*)(ws + (4u << 20) + 8192 + 4 * 65536);

  // Hypernet: generate per-sample adapter params.
  hyper_kernel<<<(A_ * D_) / 256, 256, 0, stream>>>(emb, down_w_W, down_w_b,
                                                    nullptr, ws_wd, A_ * D_);
  hyper_kernel<<<(D_ * A_) / 256, 256, 0, stream>>>(emb, up_w_W, up_w_b,
                                                    nullptr, ws_wu, D_ * A_);
  hyper_kernel<<<1, 256, 0, stream>>>(emb, down_b_W, down_b_b, ws_bdown,
                                      nullptr, A_);
  hyper_kernel<<<2, 256, 0, stream>>>(emb, up_b_W, up_b_b, ws_bup, nullptr, D_);
  hyper_kernel<<<2, 256, 0, stream>>>(emb, pre_w_W, pre_w_b, ws_prew, nullptr, D_);
  hyper_kernel<<<2, 256, 0, stream>>>(emb, pre_b_W, pre_b_b, ws_preb, nullptr, D_);
  hyper_kernel<<<2, 256, 0, stream>>>(emb, post_w_W, post_w_b, ws_postw, nullptr, D_);
  hyper_kernel<<<2, 256, 0, stream>>>(emb, post_b_W, post_b_b, ws_postb, nullptr, D_);

  // Fused adapter: one block per (s-tile, batch), 244 KB dynamic LDS.
  (void)hipFuncSetAttribute((const void*)adapter_kernel,
                            hipFuncAttributeMaxDynamicSharedMemorySize,
                            SMEM_BYTES);
  dim3 grid(S_ / TILE_S, B_);
  adapter_kernel<<<grid, 256, SMEM_BYTES, stream>>>(
      x, ws_wd, ws_wu, ws_bdown, ws_bup, ws_prew, ws_preb, ws_postw, ws_postb,
      (float*)d_out);
}
